// MossFormerM2_30416958390532
// MI455X (gfx1250) — matrix operand order repaired and checked by reading, not compiled, not run on hardware
//
#include <hip/hip_runtime.h>
#include <hip/hip_bf16.h>
#include <math.h>

// ---------------------------------------------------------------------------
// MossFormer / FLASH stack for MI455X (gfx1250, wave32, WMMA + TDM).
// All matmuls go through v_wmma_f32_16x16x32_f16; intermediates are fp16 to
// halve HBM traffic (workload is bandwidth-bound: ~1.45 TFLOP vs ~10GB moved).
// GEMM tiles are fed by the Tensor Data Mover (tensor_load_to_lds, TENSORcnt)
// where the operand is already in [rows][K] order; transposed operands fall
// back to manual LDS-transpose staging.
// ---------------------------------------------------------------------------

typedef _Float16 half_t;
typedef __attribute__((ext_vector_type(16))) _Float16 v16h;
typedef __attribute__((ext_vector_type(8)))  float    v8f;

#define BM 128
#define BN 128
#define BK 32

enum { EPI_PLAIN = 0, EPI_SILU = 1, EPI_RELU2 = 2, EPI_ATOMIC = 3 };

// ---------------- Tensor Data Mover (async tile -> LDS) --------------------
#if defined(__HIP_DEVICE_COMPILE__) && defined(__gfx1250__) && \
    __has_builtin(__builtin_amdgcn_tensor_load_to_lds)
#define HAVE_TDM 1
#else
#define HAVE_TDM 0
#endif

#if HAVE_TDM
typedef unsigned int u32x4 __attribute__((ext_vector_type(4)));
typedef int          i32x4 __attribute__((ext_vector_type(4)));
typedef int          i32x8 __attribute__((ext_vector_type(8)));

// 2D tile load: rows x rowlen fp16 elements, global row stride ld_elems,
// packed row-major into LDS at lds_off. D# built per ISA 08_async_tensor §8.
__device__ __forceinline__ void tdm_load_tile_f16(unsigned lds_off,
                                                  const half_t* gptr,
                                                  unsigned rows, unsigned rowlen,
                                                  unsigned long long ld_elems)
{
  unsigned long long ga = (unsigned long long)(size_t)gptr;
  u32x4 g0;
  g0.x = 1u;                                        // count=1 (valid), user mode
  g0.y = lds_off;                                   // lds_addr [63:32]
  g0.z = (unsigned)(ga & 0xffffffffu);              // global_addr [95:64]
  g0.w = (unsigned)((ga >> 32) & 0x01ffffffu)       // global_addr [120:96]
       | (2u << 30);                                // type=2 ("image")
  const unsigned td0 = 1u << 24, td1 = 1u << 24;    // large tensor dims: no clip
  i32x8 g1;
  g1[0] = (int)(1u << 16);                          // data_size=1 (2 bytes)
  g1[1] = (int)((td0 & 0xffffu) << 16);             // tensor_dim0[15:0]
  g1[2] = (int)(((td0 >> 16) & 0xffffu) | ((td1 & 0xffffu) << 16));
  g1[3] = (int)(((td1 >> 16) & 0xffffu) | ((rowlen & 0xffffu) << 16)); // tile_dim0
  g1[4] = (int)(rows & 0xffffu);                    // tile_dim1 (tile_dim2=0)
  g1[5] = (int)(ld_elems & 0xffffffffu);            // tensor_dim0_stride[31:0]
  g1[6] = (int)((ld_elems >> 32) & 0xffffu);        // stride[47:32], dim1_stride=0
  g1[7] = 0;
  i32x4 z4 = {0, 0, 0, 0};
#if __clang_major__ >= 23
  i32x8 z8 = {0, 0, 0, 0, 0, 0, 0, 0};
  __builtin_amdgcn_tensor_load_to_lds(g0, g1, z4, z4, z8, 0);
#else
  __builtin_amdgcn_tensor_load_to_lds(g0, g1, z4, z4, 0);
#endif
}
#endif  // HAVE_TDM

// ---- WMMA fragment loads from an LDS tile ---------------------------------
// A tile: [16 rows(M)][32 K], row stride BK. Per ISA 7.12.2 (16-bit A 16x32):
// lanes 0-15: slots 0..7 = K0..7, slots 8..15 = K16..23; lanes 16-31: +8.
__device__ __forceinline__ v16h frag_a(const half_t* tile, int lane) {
  const half_t* p = tile + (size_t)(lane & 15) * BK + ((lane >> 4) << 3);
  union { v16h v; uint4 q[2]; } u;
  u.q[0] = *(const uint4*)(p);
  u.q[1] = *(const uint4*)(p + 16);
  return u.v;
}
// B tile stored as [16 rows(N)][32 K]. Same K order as A: lanes 0-15
// slots 0..7 = K0..7, slots 8..15 = K16..23; lanes 16-31: +8.
__device__ __forceinline__ v16h frag_b(const half_t* tile, int lane) {
  const half_t* p = tile + (size_t)(lane & 15) * BK + ((lane >> 4) << 3);
  union { v16h v; uint4 q[2]; } u;
  u.q[0] = *(const uint4*)(p);
  u.q[1] = *(const uint4*)(p + 16);
  return u.v;
}

// ---- Generic batched f16 WMMA GEMM with fused epilogues -------------------
// A: memory [M][K] (transA=0) or [K][M] (transA=1), leading dim lda.
// B: memory [N][K] (transB=0) or [K][N] (transB=1), leading dim ldb.
// C: [M][N] (ldc). Requires 128|M, 128|N, 32|(K/ksplit).
__global__ __launch_bounds__(256)
void gemm_f16_wmma(const half_t* __restrict__ A, const half_t* __restrict__ B,
                   half_t* __restrict__ Ch, float* __restrict__ Cf,
                   int M, int N, int K, int lda, int ldb, int ldc,
                   long long sA, long long sB, long long sC,
                   int transA, int transB, int ksplit,
                   const float* __restrict__ bias,
                   const float* __restrict__ scale_ptr, float scale_const,
                   int mode)
{
  __shared__ half_t lA[BM * BK];
  __shared__ half_t lB[BN * BK];

  const int t    = threadIdx.x;
  const int lane = t & 31;
  const int wid  = t >> 5;
  const int wm0  = (wid >> 2) * 64;   // 2 wave rows x 64
  const int wn0  = (wid & 3) * 32;    // 4 wave cols x 32

  const int nt = blockIdx.x * BN;
  const int mt = blockIdx.y * BM;
  const int z  = blockIdx.z;
  const int batch = z / ksplit;
  const int slice = z % ksplit;
  const int kPer  = K / ksplit;
  const int kBeg  = slice * kPer;
  const int kEnd  = kBeg + kPer;

  const half_t* Ab = A + (long long)batch * sA;
  const half_t* Bb = B + (long long)batch * sB;

#if HAVE_TDM
  const unsigned ldsA_off = (unsigned)(size_t)(void*)&lA[0];
  const unsigned ldsB_off = (unsigned)(size_t)(void*)&lB[0];
#endif

  v8f acc[4][2];
  const v8f vzero = {0.f, 0.f, 0.f, 0.f, 0.f, 0.f, 0.f, 0.f};
#pragma unroll
  for (int i = 0; i < 4; ++i)
#pragma unroll
    for (int j = 0; j < 2; ++j) acc[i][j] = vzero;

  for (int k0 = kBeg; k0 < kEnd; k0 += BK) {
#if HAVE_TDM
    // ---- async DMA staging for [rows][K]-resident operands ----
    if (wid == 0) {
      if (!transA)
        tdm_load_tile_f16(ldsA_off, &Ab[(size_t)mt * lda + k0], BM, BK,
                          (unsigned long long)lda);
      if (!transB)
        tdm_load_tile_f16(ldsB_off, &Bb[(size_t)nt * ldb + k0], BN, BK,
                          (unsigned long long)ldb);
      if (!transA || !transB) __builtin_amdgcn_s_wait_tensorcnt(0);
    }
#else
    if (!transA) {
#pragma unroll
      for (int i = 0; i < 2; ++i) {
        int idx = t * 16 + i * 8;
        int r = idx >> 5, c = idx & 31;
        const half_t* gp = &Ab[(size_t)(mt + r) * lda + (k0 + c)];
        *(uint4*)&lA[idx] = *(const uint4*)gp;
        if (k0 + BK < kEnd) __builtin_prefetch(gp + BK, 0, 1);
      }
    }
    if (!transB) {
#pragma unroll
      for (int i = 0; i < 2; ++i) {
        int idx = t * 16 + i * 8;
        int r = idx >> 5, c = idx & 31;
        const half_t* gp = &Bb[(size_t)(nt + r) * ldb + (k0 + c)];
        *(uint4*)&lB[idx] = *(const uint4*)gp;
        if (k0 + BK < kEnd) __builtin_prefetch(gp + BK, 0, 1);
      }
    }
#endif
    if (transA) {  // memory [K][M]: coalesced k-rows, transpose into LDS
      int k = t >> 3, m0 = (t & 7) * 16;
      const half_t* src = &Ab[(size_t)(k0 + k) * lda + (mt + m0)];
      union { uint4 q[2]; half_t e[16]; } u;
      u.q[0] = *(const uint4*)(src);
      u.q[1] = *(const uint4*)(src + 8);
      if (k0 + BK < kEnd) __builtin_prefetch(src + (size_t)BK * lda, 0, 1);
#pragma unroll
      for (int e = 0; e < 16; ++e) lA[(m0 + e) * BK + k] = u.e[e];
    }
    if (transB) {  // memory [K][N]
      int k = t >> 3, n0 = (t & 7) * 16;
      const half_t* src = &Bb[(size_t)(k0 + k) * ldb + (nt + n0)];
      union { uint4 q[2]; half_t e[16]; } u;
      u.q[0] = *(const uint4*)(src);
      u.q[1] = *(const uint4*)(src + 8);
      if (k0 + BK < kEnd) __builtin_prefetch(src + (size_t)BK * ldb, 0, 1);
#pragma unroll
      for (int e = 0; e < 16; ++e) lB[(n0 + e) * BK + k] = u.e[e];
    }
    __syncthreads();

    v16h af[4], bf[2];
#pragma unroll
    for (int i = 0; i < 4; ++i) af[i] = frag_a(&lA[(wm0 + i * 16) * BK], lane);
#pragma unroll
    for (int j = 0; j < 2; ++j) bf[j] = frag_b(&lB[(wn0 + j * 16) * BK], lane);
#pragma unroll
    for (int i = 0; i < 4; ++i)
#pragma unroll
      for (int j = 0; j < 2; ++j)
        acc[i][j] = __builtin_amdgcn_wmma_f32_16x16x32_f16(
            false, af[i], false, bf[j], (short)0, acc[i][j], false, false);
    __syncthreads();
  }

  // ---- epilogue ----
  float s = scale_const * (scale_ptr ? *scale_ptr : 1.0f);
#pragma unroll
  for (int i = 0; i < 4; ++i)
#pragma unroll
    for (int j = 0; j < 2; ++j) {
      int col = nt + wn0 + j * 16 + (lane & 15);
      int rb  = mt + wm0 + i * 16 + ((lane >> 4) << 3);
      float bval = (bias && mode == EPI_SILU) ? bias[col] : 0.0f;
#pragma unroll
      for (int r = 0; r < 8; ++r) {
        float v = acc[i][j][r];
        long long cidx = (long long)batch * sC + (long long)(rb + r) * ldc + col;
        if (mode == EPI_SILU) {
          v = s * v + bval;
          v = v / (1.0f + __expf(-v));
          Ch[cidx] = (half_t)v;
        } else if (mode == EPI_RELU2) {
          v *= s; v = fmaxf(v, 0.0f);
          Ch[cidx] = (half_t)(v * v);
        } else if (mode == EPI_ATOMIC) {
          atomicAdd(&Cf[cidx], v * s);
        } else {
          Ch[cidx] = (half_t)(v * s);
        }
      }
    }
}

// ---- time-shift halves + RMS scale-norm (g folded into GEMM epilogue) -----
__global__ __launch_bounds__(256)
void shift_norm_kernel(const float* __restrict__ x, half_t* __restrict__ xn, int L)
{
  int tok = blockIdx.x, l = tok % L, t = threadIdx.x;
  __shared__ float red[256];
  float vals[2], ss = 0.f;
#pragma unroll
  for (int j = 0; j < 2; ++j) {
    int d = t + j * 256;
    float v;
    if (d < 256) v = (l == 0) ? 0.f : x[(long long)(tok - 1) * 512 + d];
    else         v = x[(long long)tok * 512 + d];
    vals[j] = v; ss += v * v;
  }
  red[t] = ss; __syncthreads();
  for (int o = 128; o > 0; o >>= 1) { if (t < o) red[t] += red[t + o]; __syncthreads(); }
  float n = sqrtf(red[0]) * 0.044194173824159216f;  // 512^-0.5
  float inv = 1.0f / fmaxf(n, 1e-5f);
#pragma unroll
  for (int j = 0; j < 2; ++j)
    xn[(long long)tok * 512 + t + j * 256] = (half_t)(vals[j] * inv);
}

// ---- depthwise conv (K=17) residual; optional f32 residual-accumulate -----
__global__ void dwconv_kernel(const half_t* __restrict__ h0, const float* __restrict__ kern,
                              int C, int L, long long total,
                              half_t* __restrict__ hout, float* __restrict__ xres)
{
  long long idx = (long long)blockIdx.x * blockDim.x + threadIdx.x;
  if (idx >= total) return;
  int c = (int)(idx % C);
  long long pos = idx / C;
  int l = (int)(pos % L);
  long long base = (pos - l) * C + c;
  float acc = 0.f;
#pragma unroll
  for (int tp = 0; tp < 17; ++tp) {
    int ll = l + tp - 8;
    if (0 <= ll && ll < L) acc += (float)h0[base + (long long)ll * C] * kern[c * 17 + tp];
  }
  float v = (float)h0[idx] + acc;
  if (xres) xres[idx] += v;      // final: x + (h2 + conv(h2))
  else      hout[idx] = (half_t)v;
}

// ---- qk -> 4 heads (gamma/beta) + rotary on first 32 dims -----------------
__global__ void qk4_rot_kernel(const half_t* __restrict__ qk,
                               const float* __restrict__ gamma, const float* __restrict__ beta,
                               const float* __restrict__ cosT, const float* __restrict__ sinT,
                               int L, half_t* qq, half_t* lq, half_t* qkk, half_t* lk)
{
  int tok = blockIdx.x, d = threadIdx.x, l = tok % L;
  float a = (float)qk[(long long)tok * 128 + d];
  float b = (d < 32) ? (float)qk[(long long)tok * 128 + (d ^ 1)] : 0.f;
  float c = 1.f, s = 0.f;
  if (d < 32) { int i = d >> 1; c = cosT[l * 16 + i]; s = sinT[l * 16 + i]; }
  half_t* outs[4] = {qq, lq, qkk, lk};
#pragma unroll
  for (int h = 0; h < 4; ++h) {
    float ta = a * gamma[h * 128 + d] + beta[h * 128 + d];
    float v;
    if (d < 32) {
      float tb = b * gamma[h * 128 + (d ^ 1)] + beta[h * 128 + (d ^ 1)];
      v = (d & 1) ? (ta * c + tb * s) : (ta * c - tb * s);
    } else v = ta;
    outs[h][(long long)tok * 128 + d] = (half_t)v;
  }
}

__global__ void rope_tables_kernel(float* cosT, float* sinT, int L)
{
  int idx = blockIdx.x * blockDim.x + threadIdx.x;
  if (idx >= L * 16) return;
  int l = idx >> 4, i = idx & 15;
  float inv = powf(10000.f, -(float)(2 * i) / 32.f);
  float f = (float)l * inv;
  cosT[idx] = cosf(f); sinT[idx] = sinf(f);
}

// ---- gate: att_u*v*sigmoid(att_v*u), then RMS-norm over 1024 --------------
__global__ __launch_bounds__(256)
void combine_kernel(const half_t* __restrict__ quad, const half_t* __restrict__ lin,
                    const half_t* __restrict__ h, half_t* __restrict__ outg)
{
  int tok = blockIdx.x, t = threadIdx.x;
  long long b2 = (long long)tok * 2048;
  __shared__ float red[256];
  float o[4], ss = 0.f;
#pragma unroll
  for (int j = 0; j < 4; ++j) {
    int c = t + j * 256;
    float av = (float)quad[b2 + c] + (float)lin[b2 + c];
    float au = (float)quad[b2 + 1024 + c] + (float)lin[b2 + 1024 + c];
    float v  = (float)h[b2 + c];
    float u  = (float)h[b2 + 1024 + c];
    float sg = 1.f / (1.f + __expf(-(av * u)));
    float val = au * v * sg;
    o[j] = val; ss += val * val;
  }
  red[t] = ss; __syncthreads();
  for (int off = 128; off > 0; off >>= 1) { if (t < off) red[t] += red[t + off]; __syncthreads(); }
  float n = sqrtf(red[0]) * 0.03125f;  // 1024^-0.5
  float inv = 1.f / fmaxf(n, 1e-5f);
#pragma unroll
  for (int j = 0; j < 4; ++j)
    outg[(long long)tok * 1024 + t + j * 256] = (half_t)(o[j] * inv);
}

// ---- misc helpers ---------------------------------------------------------
__global__ void transpose_cast_kernel(const float* __restrict__ W, half_t* __restrict__ out,
                                      int K, int N)  // W:[K][N] -> out:[N][K] fp16
{
  long long idx = (long long)blockIdx.x * blockDim.x + threadIdx.x;
  if (idx >= (long long)K * N) return;
  int n = (int)(idx / K), k = (int)(idx % K);
  out[idx] = (half_t)W[(long long)k * N + n];
}

__global__ void copy_f32_kernel(const float* __restrict__ a, float* __restrict__ b, long long n)
{
  long long i = (long long)blockIdx.x * blockDim.x + threadIdx.x;
  if (i < n) b[i] = a[i];
}

__global__ void zero_f32_kernel(float* __restrict__ a, long long n)
{
  long long i = (long long)blockIdx.x * blockDim.x + threadIdx.x;
  if (i < n) a[i] = 0.f;
}

__global__ void linkv_cast_kernel(const float* __restrict__ linkv, half_t* __restrict__ outT,
                                  float invL)  // [b][128][2048] -> [b][2048][128] fp16
{
  long long idx = (long long)blockIdx.x * blockDim.x + threadIdx.x;
  if (idx >= 4LL * 128 * 2048) return;
  int b = (int)(idx / (128 * 2048));
  int r = (int)(idx % (128 * 2048));
  int e = r / 128, d = r % 128;
  outT[idx] = (half_t)(linkv[(long long)b * 128 * 2048 + (long long)d * 2048 + e] * invL);
}

__global__ __launch_bounds__(256)
void layernorm_kernel(const float* __restrict__ x, const float* __restrict__ w,
                      const float* __restrict__ b, float* __restrict__ out)
{
  int tok = blockIdx.x, t = threadIdx.x;
  __shared__ float red[256];
  long long base = (long long)tok * 512;
  float v0 = x[base + t], v1 = x[base + 256 + t];
  red[t] = v0 + v1; __syncthreads();
  for (int o = 128; o > 0; o >>= 1) { if (t < o) red[t] += red[t + o]; __syncthreads(); }
  float mu = red[0] * (1.f / 512.f);
  __syncthreads();
  float d0 = v0 - mu, d1 = v1 - mu;
  red[t] = d0 * d0 + d1 * d1; __syncthreads();
  for (int o = 128; o > 0; o >>= 1) { if (t < o) red[t] += red[t + o]; __syncthreads(); }
  float inv = rsqrtf(red[0] * (1.f / 512.f) + 1e-6f);
  out[base + t]       = d0 * inv * w[t]       + b[t];
  out[base + 256 + t] = d1 * inv * w[256 + t] + b[256 + t];
}

// ---------------------------------------------------------------------------
extern "C" void kernel_launch(void* const* d_in, const int* in_sizes, int n_in,
                              void* d_out, int out_size, void* d_ws, size_t ws_size,
                              hipStream_t stream)
{
  (void)in_sizes; (void)n_in; (void)out_size; (void)ws_size;
  const int B = 4, L = 8192, BL = B * L, NL = 8;
  const float* src  = (const float*)d_in[0];
  const float* ln_w = (const float*)d_in[1 + NL * 14 + 0];
  const float* ln_b = (const float*)d_in[1 + NL * 14 + 1];
  auto P = [&](int lyr, int f) -> const float* { return (const float*)d_in[1 + lyr * 14 + f]; };
  // fields: 0 sn1, 1 Wh, 2 bh, 3 kh, 4 sn2, 5 Wqk, 6 bqk, 7 kqk,
  //         8 gamma, 9 beta, 10 sn3, 11 Wo, 12 bo, 13 ko

  size_t off = 0;
  auto alloc = [&](size_t bytes) -> void* {
    off = (off + 255) & ~(size_t)255;
    void* p = (char*)d_ws + off;
    off += bytes;
    return p;
  };

  half_t *WhT[NL], *WqkT[NL], *WoT[NL];
  for (int i = 0; i < NL; ++i) {
    WhT[i]  = (half_t*)alloc((size_t)2048 * 512 * 2);
    WqkT[i] = (half_t*)alloc((size_t)128 * 512 * 2);
    WoT[i]  = (half_t*)alloc((size_t)512 * 1024 * 2);
  }
  float*  cosT  = (float*)alloc((size_t)L * 16 * 4);
  float*  sinT  = (float*)alloc((size_t)L * 16 * 4);
  float*  xbuf  = (float*)alloc((size_t)BL * 512 * 4);
  half_t* xn    = (half_t*)alloc((size_t)BL * 512 * 2);
  half_t* h0    = (half_t*)alloc((size_t)BL * 2048 * 2);
  half_t* hbuf  = (half_t*)alloc((size_t)BL * 2048 * 2);
  half_t* qk0   = (half_t*)alloc((size_t)BL * 128 * 2);
  half_t* qkc   = (half_t*)alloc((size_t)BL * 128 * 2);
  half_t* qq    = (half_t*)alloc((size_t)BL * 128 * 2);
  half_t* lq    = (half_t*)alloc((size_t)BL * 128 * 2);
  half_t* qkk   = (half_t*)alloc((size_t)BL * 128 * 2);
  half_t* lk    = (half_t*)alloc((size_t)BL * 128 * 2);
  half_t* attn  = (half_t*)alloc((size_t)128 * 256 * 256 * 2);
  half_t* quad  = (half_t*)alloc((size_t)BL * 2048 * 2);
  float*  linkv = (float*)alloc((size_t)4 * 128 * 2048 * 4);
  half_t* linkvT= (half_t*)alloc((size_t)4 * 2048 * 128 * 2);
  half_t* lin   = (half_t*)alloc((size_t)BL * 2048 * 2);
  half_t* outg  = (half_t*)alloc((size_t)BL * 1024 * 2);
  half_t* o0    = (half_t*)alloc((size_t)BL * 512 * 2);

  dim3 blk(256);
  auto gemm = [&](const half_t* A, const half_t* Bm, half_t* Ch, float* Cf,
                  int M, int N, int K, int lda, int ldb, int ldc,
                  long long sA, long long sB, long long sC,
                  int tA, int tB, int ksplit, int batches,
                  const float* bias, const float* sp, float sc, int mode) {
    dim3 g(N / 128, M / 128, batches * ksplit);
    gemm_f16_wmma<<<g, blk, 0, stream>>>(A, Bm, Ch, Cf, M, N, K, lda, ldb, ldc,
                                         sA, sB, sC, tA, tB, ksplit, bias, sp, sc, mode);
  };
  auto ew = [&](long long n) { return dim3((unsigned)((n + 255) / 256)); };

  // ---- setup: pack weights, rotary tables, x := src ----
  for (int i = 0; i < NL; ++i) {
    transpose_cast_kernel<<<ew((long long)512 * 2048), blk, 0, stream>>>(P(i, 1),  WhT[i],  512, 2048);
    transpose_cast_kernel<<<ew((long long)512 * 128),  blk, 0, stream>>>(P(i, 5),  WqkT[i], 512, 128);
    transpose_cast_kernel<<<ew((long long)1024 * 512), blk, 0, stream>>>(P(i, 11), WoT[i],  1024, 512);
  }
  rope_tables_kernel<<<ew((long long)L * 16), blk, 0, stream>>>(cosT, sinT, L);
  copy_f32_kernel<<<ew((long long)BL * 512), blk, 0, stream>>>(src, xbuf, (long long)BL * 512);

  for (int lyr = 0; lyr < NL; ++lyr) {
    // 1) xs = shift-halves(x); xn = xs / max(rms, eps)   (g folded into GEMMs)
    shift_norm_kernel<<<BL, blk, 0, stream>>>(xbuf, xn, L);

    // 2) h0 = silu(sn1 * xn@Wh + bh); qk0 = silu(sn2 * xn@Wqk + bqk)
    gemm(xn, WhT[lyr],  h0,  nullptr, BL, 2048, 512, 512, 512, 2048, 0, 0, 0,
         0, 0, 1, 1, P(lyr, 2), P(lyr, 0), 1.0f, EPI_SILU);
    gemm(xn, WqkT[lyr], qk0, nullptr, BL, 128,  512, 512, 512, 128,  0, 0, 0,
         0, 0, 1, 1, P(lyr, 6), P(lyr, 4), 1.0f, EPI_SILU);

    // 3) depthwise conv-17 residual
    dwconv_kernel<<<ew((long long)BL * 2048), blk, 0, stream>>>(h0, P(lyr, 3), 2048, L,
                                                                (long long)BL * 2048, hbuf, nullptr);
    dwconv_kernel<<<ew((long long)BL * 128), blk, 0, stream>>>(qk0, P(lyr, 7), 128, L,
                                                               (long long)BL * 128, qkc, nullptr);

    // 4) 4 heads + rotary
    qk4_rot_kernel<<<BL, dim3(128), 0, stream>>>(qkc, P(lyr, 8), P(lyr, 9), cosT, sinT, L,
                                                 qq, lq, qkk, lk);

    // 5) quad attention per (b, group): attn = relu(qq@qk^T/256)^2  [128 batches]
    gemm(qq, qkk, attn, nullptr, 256, 256, 128, 128, 128, 256,
         256LL * 128, 256LL * 128, 256LL * 256, 0, 0, 1, 128,
         nullptr, nullptr, 1.0f / 256.0f, EPI_RELU2);
    //    quad = attn @ [v:u]   (B memory is [pos][chan] => transB stage)
    gemm(attn, hbuf, quad, nullptr, 256, 2048, 256, 256, 2048, 2048,
         256LL * 256, 256LL * 2048, 256LL * 2048, 0, 1, 1, 128,
         nullptr, nullptr, 1.0f, EPI_PLAIN);

    // 6) linear attention: linkv[b] = lk^T @ [v:u]  (K=8192, 8-way K-split + f32 atomics)
    zero_f32_kernel<<<ew(4LL * 128 * 2048), blk, 0, stream>>>(linkv, 4LL * 128 * 2048);
    gemm(lk, hbuf, nullptr, linkv, 128, 2048, 8192, 128, 2048, 2048,
         8192LL * 128, 8192LL * 2048, 128LL * 2048, 1, 1, 8, 4,
         nullptr, nullptr, 1.0f, EPI_ATOMIC);
    linkv_cast_kernel<<<ew(4LL * 128 * 2048), blk, 0, stream>>>(linkv, linkvT, 1.0f / (float)L);
    //    lin[b] = lq @ linkv[b]
    gemm(lq, linkvT, lin, nullptr, 8192, 2048, 128, 128, 128, 2048,
         8192LL * 128, 2048LL * 128, 8192LL * 2048, 0, 0, 1, 4,
         nullptr, nullptr, 1.0f, EPI_PLAIN);

    // 7) gate + RMS-norm (sn3 folded into Wo GEMM)
    combine_kernel<<<BL, blk, 0, stream>>>(quad, lin, hbuf, outg);

    // 8) o0 = silu(sn3 * outg@Wo + bo); x += o0 + conv(o0)
    gemm(outg, WoT[lyr], o0, nullptr, BL, 512, 1024, 1024, 1024, 512, 0, 0, 0,
         0, 0, 1, 1, P(lyr, 12), P(lyr, 10), 1.0f, EPI_SILU);
    dwconv_kernel<<<ew((long long)BL * 512), blk, 0, stream>>>(o0, P(lyr, 13), 512, L,
                                                               (long long)BL * 512, nullptr, xbuf);
  }

  // final LayerNorm -> f32 output
  layernorm_kernel<<<BL, blk, 0, stream>>>(xbuf, ln_w, ln_b, (float*)d_out);
}
